// SAGEConvWithEdgeAttrAndEmbedding_24051816857685
// MI455X (gfx1250) — matrix operand order repaired and checked by reading, not compiled, not run on hardware
//
#include <hip/hip_runtime.h>
#include <hip/hip_bf16.h>
#include <math.h>

// ---------------------------------------------------------------------------
// SAGEConv + edge-attr gating + BN, fused for MI455X (gfx1250, wave32, WMMA).
// Heavy GEMMs run on v_wmma_f32_16x16x32_bf16 (f32 accumulate). Weights are
// pre-packed into the per-lane WMMA B-fragment layout. Both the 96 KB gate
// fragment buffer (edge kernel) and the 64 KB [W_l;W_r] fragment buffer
// (node kernel) are staged into LDS once per block by the Tensor Data Mover
// (tensor_load_to_lds + s_wait_tensorcnt); GEMMs read B-fragments from LDS.
// x/out (25.6 MB each) stay L2-resident (192 MB L2).
// ---------------------------------------------------------------------------

typedef __attribute__((ext_vector_type(16))) __bf16 v16bf;
typedef __attribute__((ext_vector_type(8)))  __bf16 v8bf;
typedef __attribute__((ext_vector_type(8)))  float  v8f;
typedef __attribute__((ext_vector_type(4)))  unsigned int u32x4;
typedef __attribute__((ext_vector_type(8)))  int          i32x8;
typedef __attribute__((ext_vector_type(4)))  int          i32x4;

#define WMMA_BF16(a, b, c) \
    __builtin_amdgcn_wmma_f32_16x16x32_bf16(false, (a), false, (b), (short)0, (c), false, false)

constexpr int N_NODES = 50000;
constexpr int N_EDGES = 800000;
constexpr int DIN  = 128;
constexpr int DOUT = 128;
constexpr int DEA  = 32;
constexpr int DEE  = 16;

// Packed-weight buffer layout (units: bf16 elements; every region is a whole
// number of 512-element (32 lane x 16 elem) WMMA B tiles).
constexpr int PK_EMB   = 0;              // 1 tile   : W_emb   [32 x 16]
constexpr int PK_TEMB  = 512;            // 8 tiles  : W_temb  [16(pad32) x 128]
constexpr int PK_TATTR = 4608;           // 8 tiles  : W_tattr [32 x 128]
constexpr int PK_GATE  = 8704;           // 96 tiles : W_gate  [384 x 128]
constexpr int PK_LR    = 57856;          // 64 tiles : [W_l;W_r] [256 x 128]
constexpr int PK_TOTAL = 90624;

constexpr int GATE_BYTES = 96 * 512 * 2;            // 98304 B staged via TDM
constexpr int COMB_BYTES = 4 * 16 * 384 * 2;        // 49152 B
constexpr int EMB_BYTES  = 4 * 16 * 16 * 4;         //  4096 B
constexpr int EDGE_SMEM  = GATE_BYTES + COMB_BYTES + EMB_BYTES;  // 151552 B
constexpr int LR_BYTES   = 64 * 512 * 2;            // 65536 B staged via TDM
constexpr int NODE_SMEM  = LR_BYTES;

// ---------------------------------------------------------------------------
// Tensor Data Mover: 1-D copy of n_qw 8-byte elements into LDS.
// D# built per CDNA5 ISA (group0 128b, group1 256b).
__device__ inline void tdm_stage(const void* src, unsigned lds_off, unsigned n_qw) {
    unsigned long long ga = (unsigned long long)src;
    u32x4 g0;
    g0[0] = 1u;                                   // count=1, user mode, no gather
    g0[1] = lds_off;                              // lds_addr (bytes)
    g0[2] = (unsigned)(ga & 0xFFFFFFFFu);         // global_addr[31:0]
    g0[3] = (unsigned)((ga >> 32) & 0x1FFFFFFu)   // global_addr[56:32]
          | (2u << 30);                           // type=2 ("image")
    i32x8 g1;
    g1[0] = (int)(3u << 16);                      // workgroup_mask=0, data_size=3 (8B)
    g1[1] = (int)((n_qw & 0xFFFFu) << 16);        // tensor_dim0[15:0]   (bits 63:48)
    g1[2] = (int)((n_qw >> 16) | (1u << 16));     // tensor_dim0[31:16], tensor_dim1=1
    g1[3] = (int)((n_qw & 0xFFFFu) << 16);        // tile_dim0           (bits 127:112)
    g1[4] = 0;                                    // tile_dim1=0, tile_dim2=0
    g1[5] = (int)n_qw;                            // tensor_dim0_stride[31:0]
    g1[6] = 0;
    g1[7] = 0;
    i32x4 gz = {0, 0, 0, 0};
#if defined(__clang_major__) && __clang_major__ >= 23
    i32x8 gz8 = {0, 0, 0, 0, 0, 0, 0, 0};
    __builtin_amdgcn_tensor_load_to_lds(g0, g1, gz, gz, gz8, 0);
#else
    __builtin_amdgcn_tensor_load_to_lds(g0, g1, gz, gz, 0);
#endif
}

// ---------------------------------------------------------------------------
__global__ void init_zero_kernel(float* agg, float* cnt, float* bn) {
    size_t i = (size_t)blockIdx.x * blockDim.x + threadIdx.x;
    if (i < (size_t)N_NODES * DIN) agg[i] = 0.0f;
    if (i < (size_t)N_NODES)       cnt[i] = 0.0f;
    if (i < 256)                   bn[i]  = 0.0f;
}

// Pack weights (fp32, row-major [K, N]) into bf16 WMMA B-fragments.
// h = lane / 16: K = 8h + j (j < 8), K = 16 + 8h + (j - 8) (j >= 8), N = lane % 16.
__global__ void pack_weights_kernel(const float* __restrict__ W_emb,
                                    const float* __restrict__ W_temb,
                                    const float* __restrict__ W_tattr,
                                    const float* __restrict__ W_gate,
                                    const float* __restrict__ W_l,
                                    const float* __restrict__ W_r,
                                    __bf16* __restrict__ pk) {
    int i = blockIdx.x * blockDim.x + threadIdx.x;
    if (i >= PK_TOTAL) return;
    int j    = i & 15;
    int lane = (i >> 4) & 31;
    int tile = i >> 9;
    int k_in, n_in;
    if (lane < 16) { k_in = j + ((j >= 8) ? 8 : 0);     n_in = lane;      }
    else           { k_in = 8 + j + ((j >= 8) ? 8 : 0); n_in = lane - 16; }

    float v = 0.0f;
    if (i < PK_TEMB) {                       // W_emb: K 0..31, N 0..15
        v = W_emb[k_in * DEE + n_in];
    } else if (i < PK_TATTR) {               // W_temb: K 0..15 valid, zero-pad
        int nt = tile - 1;
        if (k_in < DEE) v = W_temb[k_in * DOUT + nt * 16 + n_in];
    } else if (i < PK_GATE) {                // W_tattr: K 0..31
        int nt = tile - 9;
        v = W_tattr[k_in * DOUT + nt * 16 + n_in];
    } else if (i < PK_LR) {                  // W_gate: kt 0..11 x nt 0..7
        int t = tile - 17, kt = t >> 3, nt = t & 7;
        v = W_gate[(kt * 32 + k_in) * DOUT + nt * 16 + n_in];
    } else {                                 // [W_l ; W_r]: kt 0..7 x nt 0..7
        int t = tile - 113, kt = t >> 3, nt = t & 7;
        int K = kt * 32 + k_in;
        v = (K < DIN) ? W_l[K * DOUT + nt * 16 + n_in]
                      : W_r[(K - DIN) * DOUT + nt * 16 + n_in];
    }
    pk[i] = (__bf16)v;
}

// Scatter-mean numerator/denominator: agg[col] += x[row], cnt[col] += 1.
__global__ void scatter_mean_kernel(const float* __restrict__ x,
                                    const int* __restrict__ row,
                                    const int* __restrict__ col,
                                    float* __restrict__ agg,
                                    float* __restrict__ cnt) {
    size_t idx = (size_t)blockIdx.x * blockDim.x + threadIdx.x;
    if (idx >= (size_t)N_EDGES * 32) return;
    int e = (int)(idx >> 5);
    int part = (int)(idx & 31);
    int r = row[e], c = col[e];
    const float4 v = ((const float4*)(x + (size_t)r * DIN))[part];
    float* dst = agg + (size_t)c * DIN + part * 4;
    atomicAdd(dst + 0, v.x);
    atomicAdd(dst + 1, v.y);
    atomicAdd(dst + 2, v.z);
    atomicAdd(dst + 3, v.w);
    if (part == 0) atomicAdd(&cnt[c], 1.0f);
}

// Node GEMM: out = [agg/cnt | x](16x256) @ [W_l;W_r](256x128) + b_l.
// B-fragments staged into LDS by the TDM once per block; all 4 waves read
// them with ds_read_b128 instead of latency-bound L2 loads.
__global__ __launch_bounds__(128) void node_gemm_kernel(
        const float* __restrict__ x, const float* __restrict__ agg,
        const float* __restrict__ cnt, const __bf16* __restrict__ pk,
        const float* __restrict__ b_l,
        float* __restrict__ out0, float* __restrict__ out_acc) {
    extern __shared__ __align__(16) char nsmem[];
    __bf16* lrS = (__bf16*)nsmem;                                    // 64 KB

    int wave = threadIdx.x >> 5, lane = threadIdx.x & 31;
    int tile = blockIdx.x * 4 + wave;

    // Stage [W_l;W_r] fragments (8192 qwords) via the Tensor Data Mover.
    if (wave == 0) {
        tdm_stage(pk + PK_LR, 0u, 8192u);
        __builtin_amdgcn_s_wait_tensorcnt(0);
    }
    __syncthreads();
    if (tile >= N_NODES / 16) return;       // after the barrier: all waves sync

    int m0 = tile * 16;
    int arow  = lane & 15;
    int kbase = (lane < 16) ? 0 : 8;
    int ncol  = lane & 15;
    int mhi   = (lane >= 16) ? 8 : 0;
    int r = m0 + arow;
    float inv = 1.0f / fmaxf(cnt[r], 1.0f);

    v8f z;
    #pragma unroll
    for (int q = 0; q < 8; ++q) z[q] = 0.0f;
    v8f acc[8];
    #pragma unroll
    for (int nt = 0; nt < 8; ++nt) acc[nt] = z;

    const v16bf* lrB = (const v16bf*)lrS;
    for (int kt = 0; kt < 8; ++kt) {
        const float* src;
        float scale;
        if (kt < 4) { src = agg + (size_t)r * DIN + kt * 32;       scale = inv;  }
        else        { src = x   + (size_t)r * DIN + (kt - 4) * 32; scale = 1.0f; }
        v16bf a;
        #pragma unroll
        for (int j = 0; j < 8; ++j) {
            a[j]     = (__bf16)(src[kbase + j] * scale);
            a[8 + j] = (__bf16)(src[kbase + 16 + j] * scale);
        }
        #pragma unroll
        for (int nt = 0; nt < 8; ++nt) {
            v16bf b = lrB[(kt * 8 + nt) * 32 + lane];
            acc[nt] = WMMA_BF16(a, b, acc[nt]);
        }
    }
    #pragma unroll
    for (int nt = 0; nt < 8; ++nt) {
        float bias = b_l[nt * 16 + ncol];
        #pragma unroll
        for (int v = 0; v < 8; ++v) {
            size_t idx = (size_t)(m0 + v + mhi) * DOUT + nt * 16 + ncol;
            float o = acc[nt][v] + bias;
            out0[idx]    = o;
            out_acc[idx] = o;
        }
    }
}

// Fused per-edge pipeline: emb -> t_attr/t -> gate GEMM -> scatter gate*t.
// Dynamic LDS: [gate B-frags 96K | comb 48K | emb 4K]. Gate weights arrive
// via the Tensor Data Mover (one tensor_load_to_lds per block).
__global__ __launch_bounds__(128) void edge_fused_kernel(
        const float* __restrict__ edge_attr, const int* __restrict__ col_idx,
        const float* __restrict__ out0, float* __restrict__ out_acc,
        const __bf16* __restrict__ pk,
        const float* __restrict__ b_emb,  const float* __restrict__ b_temb,
        const float* __restrict__ b_tattr, const float* __restrict__ b_gate) {
    extern __shared__ __align__(16) char smem[];
    __bf16* gateS = (__bf16*)smem;                                   // 96 KB
    __bf16 (*combS)[16][384] = (__bf16(*)[16][384])(smem + GATE_BYTES);
    float  (*embS)[16][16]   = (float (*)[16][16])(smem + GATE_BYTES + COMB_BYTES);

    int wave = threadIdx.x >> 5, lane = threadIdx.x & 31;
    size_t e0 = ((size_t)blockIdx.x * 4 + wave) * 16;
    int arow  = lane & 15;
    int kbase = (lane < 16) ? 0 : 8;
    int ncol  = lane & 15;
    int mhi   = (lane >= 16) ? 8 : 0;

    // Kick off the TDM copy of the gate fragments (wave 0 only; EXEC ignored,
    // dynamic LDS starts at offset 0 of this block's allocation).
    if (wave == 0) tdm_stage(pk + PK_GATE, 0u, 12288u);

    v8f z;
    #pragma unroll
    for (int q = 0; q < 8; ++q) z[q] = 0.0f;

    // A-fragment of edge_attr rows e0..e0+15 (16x32 bf16), built in registers.
    const float* ea = edge_attr + (e0 + arow) * DEA;
    v16bf a_ea;
    #pragma unroll
    for (int j = 0; j < 8; ++j) {
        a_ea[j]     = (__bf16)ea[kbase + j];
        a_ea[8 + j] = (__bf16)ea[kbase + 16 + j];
    }
    if (e0 + 64 + arow < N_EDGES)                 // prefetch next block's tile
        __builtin_prefetch(edge_attr + (e0 + 64 + arow) * DEA, 0, 1);

    // emb = relu(ea @ W_emb + b_emb): one WMMA, stage to LDS for re-layout.
    {
        v16bf b = ((const v16bf*)(pk + PK_EMB))[lane];
        v8f acc_e = WMMA_BF16(a_ea, b, z);
        float be = b_emb[ncol];
        #pragma unroll
        for (int v = 0; v < 8; ++v)
            embS[wave][v + mhi][ncol] = fmaxf(acc_e[v] + be, 0.0f);
    }

    // Gather out0[col] rows into comb cols [0,128).
    for (int rr = 0; rr < 16; ++rr) {
        int c = col_idx[e0 + rr];
        const float4 vv = ((const float4*)(out0 + (size_t)c * DOUT))[lane];
        combS[wave][rr][lane * 4 + 0] = (__bf16)vv.x;
        combS[wave][rr][lane * 4 + 1] = (__bf16)vv.y;
        combS[wave][rr][lane * 4 + 2] = (__bf16)vv.z;
        combS[wave][rr][lane * 4 + 3] = (__bf16)vv.w;
    }
    __syncthreads();

    // A-fragment of emb (K=0..15 valid, upper half zero-padded).
    v16bf a_emb;
    #pragma unroll
    for (int j = 0; j < 8; ++j) {
        a_emb[j]     = (__bf16)embS[wave][arow][kbase + j];
        a_emb[8 + j] = (__bf16)0.0f;
    }

    // t_attr = ea@W_tattr + b_tattr ; t = emb@W_temb + b_temb + t_attr.
    float t_reg[8][8];
    #pragma unroll
    for (int nt = 0; nt < 8; ++nt) {
        v16bf bta = ((const v16bf*)(pk + PK_TATTR))[nt * 32 + lane];
        v8f acc_t = WMMA_BF16(a_ea, bta, z);              // ea @ W_tattr
        v16bf bte = ((const v16bf*)(pk + PK_TEMB))[nt * 32 + lane];
        v8f acc_f = WMMA_BF16(a_emb, bte, acc_t);         // + emb @ W_temb
        float b_ta = b_tattr[nt * 16 + ncol];
        float b_te = b_temb[nt * 16 + ncol];
        #pragma unroll
        for (int v = 0; v < 8; ++v) {
            float ta = acc_t[v] + b_ta;
            float tt = acc_f[v] + b_ta + b_te;
            t_reg[nt][v] = tt;
            combS[wave][v + mhi][128 + nt * 16 + ncol] = (__bf16)tt;
            combS[wave][v + mhi][256 + nt * 16 + ncol] = (__bf16)ta;
        }
    }
    // Ensure the TDM transfer has landed before anyone reads gateS.
    if (wave == 0) __builtin_amdgcn_s_wait_tensorcnt(0);
    __syncthreads();

    // Gate GEMM: [16x384] @ W_gate[384x128]; B-fragments from LDS (ds_read).
    v8f g[8];
    #pragma unroll
    for (int nt = 0; nt < 8; ++nt) g[nt] = z;
    const v16bf* gS = (const v16bf*)gateS;
    for (int kt = 0; kt < 12; ++kt) {
        const __bf16* rp = &combS[wave][arow][kt * 32 + kbase];
        v8bf lo = *(const v8bf*)rp;
        v8bf hi = *(const v8bf*)(rp + 16);
        v16bf a;
        #pragma unroll
        for (int j = 0; j < 8; ++j) { a[j] = lo[j]; a[8 + j] = hi[j]; }
        #pragma unroll
        for (int nt = 0; nt < 8; ++nt) {
            v16bf b = gS[(kt * 8 + nt) * 32 + lane];
            g[nt] = WMMA_BF16(a, b, g[nt]);
        }
    }

    // gate = sigmoid(g + b_gate); scatter-add gate * t into out_acc[col].
    #pragma unroll
    for (int nt = 0; nt < 8; ++nt) {
        float bg = b_gate[nt * 16 + ncol];
        #pragma unroll
        for (int v = 0; v < 8; ++v) {
            float gate = 1.0f / (1.0f + __expf(-(g[nt][v] + bg)));
            float val  = gate * t_reg[nt][v];
            int c = col_idx[e0 + v + mhi];
            atomicAdd(out_acc + (size_t)c * DOUT + nt * 16 + ncol, val);
        }
    }
}

// Column-wise BN partial sums: 128 threads (one per feature), grid-stride rows.
__global__ void bn_stats_kernel(const float* __restrict__ out_acc,
                                float* __restrict__ bn_sum,
                                float* __restrict__ bn_sq) {
    int c = threadIdx.x;
    float s = 0.0f, sq = 0.0f;
    for (int r = blockIdx.x; r < N_NODES; r += gridDim.x) {
        float v = out_acc[(size_t)r * DOUT + c];
        s += v;
        sq += v * v;
    }
    atomicAdd(&bn_sum[c], s);
    atomicAdd(&bn_sq[c], sq);
}

__global__ void bn_finalize_kernel(const float* __restrict__ out_acc,
                                   const float* __restrict__ bn_sum,
                                   const float* __restrict__ bn_sq,
                                   const float* __restrict__ gamma,
                                   const float* __restrict__ beta,
                                   float* __restrict__ out) {
    size_t i = (size_t)blockIdx.x * blockDim.x + threadIdx.x;
    if (i >= (size_t)N_NODES * DOUT) return;
    int c = (int)(i & 127);
    float mu  = bn_sum[c] * (1.0f / N_NODES);
    float var = bn_sq[c] * (1.0f / N_NODES) - mu * mu;
    float v = (out_acc[i] - mu) * rsqrtf(var + 1e-5f) * gamma[c] + beta[c];
    out[i] = fmaxf(2.0f * v, 0.0f);
}

// ---------------------------------------------------------------------------
extern "C" void kernel_launch(void* const* d_in, const int* in_sizes, int n_in,
                              void* d_out, int out_size, void* d_ws, size_t ws_size,
                              hipStream_t stream) {
    const float* x         = (const float*)d_in[0];
    const float* edge_attr = (const float*)d_in[1];
    const float* W_emb     = (const float*)d_in[2];
    const float* b_emb     = (const float*)d_in[3];
    const float* W_temb    = (const float*)d_in[4];
    const float* b_temb    = (const float*)d_in[5];
    const float* W_tattr   = (const float*)d_in[6];
    const float* b_tattr   = (const float*)d_in[7];
    const float* W_gate    = (const float*)d_in[8];
    const float* b_gate    = (const float*)d_in[9];
    const float* W_l       = (const float*)d_in[10];
    const float* b_l       = (const float*)d_in[11];
    const float* W_r       = (const float*)d_in[12];
    const float* gamma     = (const float*)d_in[13];
    const float* beta      = (const float*)d_in[14];
    const int*   row       = (const int*)d_in[15];           // edge_index[0]
    const int*   col       = (const int*)d_in[15] + N_EDGES; // edge_index[1]
    float* out = (float*)d_out;

    // Workspace carve-up (256B aligned regions).
    char* w = (char*)d_ws;
    auto carve = [&](size_t bytes) {
        char* p = w;
        w += (bytes + 255) & ~(size_t)255;
        return p;
    };
    float*  agg     = (float*) carve((size_t)N_NODES * DIN * sizeof(float));
    float*  cnt     = (float*) carve((size_t)N_NODES * sizeof(float));
    float*  out0    = (float*) carve((size_t)N_NODES * DOUT * sizeof(float));
    float*  out_acc = (float*) carve((size_t)N_NODES * DOUT * sizeof(float));
    float*  bn      = (float*) carve(256 * sizeof(float));   // [sum(128) | sq(128)]
    __bf16* pk      = (__bf16*)carve((size_t)PK_TOTAL * sizeof(__bf16));
    float* bn_sum = bn;
    float* bn_sq  = bn + 128;

    // 1) zero accumulators
    {
        size_t total = (size_t)N_NODES * DIN;
        init_zero_kernel<<<(unsigned)((total + 255) / 256), 256, 0, stream>>>(agg, cnt, bn);
    }
    // 2) pack weights to bf16 WMMA B-fragments
    pack_weights_kernel<<<(PK_TOTAL + 255) / 256, 256, 0, stream>>>(
        W_emb, W_temb, W_tattr, W_gate, W_l, W_r, pk);
    // 3) scatter-mean numerator/denominator
    {
        size_t total = (size_t)N_EDGES * 32;
        scatter_mean_kernel<<<(unsigned)((total + 255) / 256), 256, 0, stream>>>(
            x, row, col, agg, cnt);
    }
    // 4) node GEMM (WMMA, TDM-staged B): out0 = [agg/cnt|x] @ [W_l;W_r] + b_l
    {
        int tiles  = N_NODES / 16;                 // 3125
        int blocks = (tiles + 3) / 4;              // 4 waves/block
        node_gemm_kernel<<<blocks, 128, NODE_SMEM, stream>>>(
            x, agg, cnt, pk, b_l, out0, out_acc);
    }
    // 5) fused per-edge pipeline (WMMA + TDM-staged gate weights)
    {
        int blocks = N_EDGES / 64;                 // 12500, 16 edges/wave
        edge_fused_kernel<<<blocks, 128, EDGE_SMEM, stream>>>(
            edge_attr, col, out0, out_acc, pk, b_emb, b_temb, b_tattr, b_gate);
    }
    // 6) BN batch stats
    bn_stats_kernel<<<256, 128, 0, stream>>>(out_acc, bn_sum, bn_sq);
    // 7) normalize + relu(2*out)
    {
        size_t total = (size_t)N_NODES * DOUT;
        bn_finalize_kernel<<<(unsigned)((total + 255) / 256), 256, 0, stream>>>(
            out_acc, bn_sum, bn_sq, gamma, beta, out);
    }
}
